// GLALayer_66383014527086
// MI455X (gfx1250) — matrix operand order repaired and checked by reading, not compiled, not run on hardware
//
#include <hip/hip_runtime.h>
#include <hip/hip_bf16.h>
#include <stdint.h>

#define D_MODEL 1024
#define HS      1024           // N_HEADS * D_STATE
#define NGATE   2048
#define N1      3072           // combined [forget | input | value] output columns
#define BATCH   4
#define SEQ     4096
#define MROWS   (BATCH * SEQ)  // 16384
#define NC      32             // scan chunks
#define CL      (SEQ / NC)     // 128 steps per chunk
#define NLANES  (BATCH * HS)   // 4096 independent scan lanes

#define BLOCK_M 128            // 8 waves x 16 rows
#define BLOCK_N 64
#define KSTEP   32

typedef __attribute__((ext_vector_type(16))) __bf16 v16bf;
typedef __attribute__((ext_vector_type(8)))  __bf16 v8bf;
typedef __attribute__((ext_vector_type(8)))  float  v8f;
typedef __attribute__((ext_vector_type(4)))  int    v4i;
typedef __attribute__((address_space(1))) v4i* as1_v4i;
typedef __attribute__((address_space(3))) v4i* as3_v4i;

#if __has_builtin(__builtin_amdgcn_global_load_async_to_lds_b128)
#define ASYNC_OK 1
#else
#define ASYNC_OK 0
#endif

__device__ __forceinline__ unsigned short f2bf(float f) {
  unsigned u = __builtin_bit_cast(unsigned, f);
  u += 0x7fffu + ((u >> 16) & 1u);          // round-to-nearest-even
  return (unsigned short)(u >> 16);
}
__device__ __forceinline__ float bf2f(unsigned short h) {
  unsigned u = ((unsigned)h) << 16;
  return __builtin_bit_cast(float, u);
}
__device__ __forceinline__ float sigmoidf_fast(float x) {
  return 1.0f / (1.0f + __expf(-x));
}

// 16-byte global -> LDS copy: async on gfx1250 if the builtin exists.
__device__ __forceinline__ void copy16_to_lds(const unsigned short* g, unsigned short* l) {
#if ASYNC_OK
  __builtin_amdgcn_global_load_async_to_lds_b128((as1_v4i)g, (as3_v4i)l, 0, 0);
#else
  *(v8bf*)l = *(const v8bf*)g;
#endif
}

__device__ __forceinline__ void wait_async_3() {
#if ASYNC_OK
#if __has_builtin(__builtin_amdgcn_s_wait_asynccnt)
  __builtin_amdgcn_s_wait_asynccnt(3);
#else
  asm volatile("s_wait_asynccnt 0x3" ::: "memory");
#endif
#endif
}
__device__ __forceinline__ void wait_async_0() {
#if ASYNC_OK
#if __has_builtin(__builtin_amdgcn_s_wait_asynccnt)
  __builtin_amdgcn_s_wait_asynccnt(0);
#else
  asm volatile("s_wait_asynccnt 0x0" ::: "memory");
#endif
#endif
}

// ---------------------------------------------------------------- convert
__global__ void f32_to_bf16_kernel(const float* __restrict__ src,
                                   unsigned short* __restrict__ dst, int n) {
  int i = blockIdx.x * blockDim.x + threadIdx.x;
  if (i < n) dst[i] = f2bf(src[i]);
}

// ---------------------------------------------------------------- WMMA GEMM
// C[M,N] = A[M,K] * W[N,K]^T ; bf16 in, f32 accumulate.
// Block: 256 threads / 8 waves own a 128x64 tile. A (8KB) and B (4KB) tiles
// are double-buffered in LDS, filled by global_load_async_to_lds_b128, with
// s_wait_asynccnt pipelining one K-step ahead. Each wave: 16(M)x64(N),
// 4x v_wmma_f32_16x16x32_bf16 per K-step, fragments read via ds_load_b128.
// MODE 0: plain f32 store. MODE 1: fused gate epilogue (sigmoid / routing).
template <int MODE>
__global__ __launch_bounds__(256)
void gla_gemm_bf16(const __bf16* __restrict__ Abuf,
                   const __bf16* __restrict__ Wbuf,
                   int Ncols,
                   float* __restrict__ outF,
                   const float* __restrict__ gate_b,
                   unsigned short* __restrict__ outA,
                   unsigned short* __restrict__ outI,
                   unsigned short* __restrict__ outV) {
  __shared__ unsigned short sA[2][BLOCK_M * KSTEP];   // 8KB per buffer
  __shared__ unsigned short sB[2][BLOCK_N * KSTEP];   // 4KB per buffer

  const int tid  = threadIdx.x;
  const int lane = tid & 31;
  const int l15  = lane & 15;
  const int half = lane >> 4;
  const int wv   = tid >> 5;                  // wave 0..7 -> M sub-tile
  const int nT   = Ncols >> 6;
  const int mBlk = blockIdx.x / nT;
  const int nBlk = blockIdx.x - mBlk * nT;
  const int mBase = mBlk * BLOCK_M;
  const int nBase = nBlk * BLOCK_N;

  const __bf16* Ablk = Abuf + (size_t)mBase * D_MODEL;
  const __bf16* Bblk = Wbuf + (size_t)nBase * D_MODEL;

  // staging map: 16B chunks. A: 512 chunks (2/thread), B: 256 chunks (1/thread)
  const int ar0 = tid >> 2,         aq0 = tid & 3;          // chunk tid
  const int ar1 = (tid + 256) >> 2, aq1 = (tid + 256) & 3;  // chunk tid+256
  const int bc  = tid >> 2,         bq  = tid & 3;

  auto stageAB = [&](int buf, int ks) {
    copy16_to_lds((const unsigned short*)(Ablk + (size_t)ar0 * D_MODEL + ks + aq0 * 8),
                  &sA[buf][ar0 * KSTEP + aq0 * 8]);
    copy16_to_lds((const unsigned short*)(Ablk + (size_t)ar1 * D_MODEL + ks + aq1 * 8),
                  &sA[buf][ar1 * KSTEP + aq1 * 8]);
    copy16_to_lds((const unsigned short*)(Bblk + (size_t)bc * D_MODEL + ks + bq * 8),
                  &sB[buf][bc * KSTEP + bq * 8]);
  };

  v8f c0 = {}, c1 = {}, c2 = {}, c3 = {};
  stageAB(0, 0);                                   // prologue
  for (int ks = 0; ks < D_MODEL; ks += KSTEP) {
    const int cur = (ks >> 5) & 1;
    if (ks + KSTEP < D_MODEL) {
      stageAB(cur ^ 1, ks + KSTEP);                // prefetch next buffer
      wait_async_3();                              // current buffer landed
    } else {
      wait_async_0();
    }
    __syncthreads();                               // visible to all waves

    // A frag: lane l<16 -> row wv*16+l, K {0..15}; lanes 16..31 K {16..31}
    const unsigned short* aRow = &sA[cur][(wv * 16 + l15) * KSTEP + half * 16];
    v8bf alo = *(const v8bf*)(aRow);
    v8bf ahi = *(const v8bf*)(aRow + 8);
    v16bf a = __builtin_shufflevector(alo, ahi,
                0,1,2,3,4,5,6,7,8,9,10,11,12,13,14,15);
    // B frag: lane holds one output column, 16 contiguous K per half
    v16bf b0 = *(const v16bf*)&sB[cur][(l15 +  0) * KSTEP + half * 16];
    v16bf b1 = *(const v16bf*)&sB[cur][(l15 + 16) * KSTEP + half * 16];
    v16bf b2 = *(const v16bf*)&sB[cur][(l15 + 32) * KSTEP + half * 16];
    v16bf b3 = *(const v16bf*)&sB[cur][(l15 + 48) * KSTEP + half * 16];

    c0 = __builtin_amdgcn_wmma_f32_16x16x32_bf16(false, a, false, b0, (short)0, c0, false, false);
    c1 = __builtin_amdgcn_wmma_f32_16x16x32_bf16(false, a, false, b1, (short)0, c1, false, false);
    c2 = __builtin_amdgcn_wmma_f32_16x16x32_bf16(false, a, false, b2, (short)0, c2, false, false);
    c3 = __builtin_amdgcn_wmma_f32_16x16x32_bf16(false, a, false, b3, (short)0, c3, false, false);

    __syncthreads();                               // done reading cur before rewrite
  }

  v8f accs[4] = {c0, c1, c2, c3};
  const int rowBase = mBase + wv * 16 + half * 8;
  // C/D layout: VGPR e -> row rowBase + e ; col nBase + f*16 + l15.
  #pragma unroll
  for (int f = 0; f < 4; ++f) {
    const int col = nBase + f * 16 + l15;
    float bias = 0.0f;
    if (MODE == 1 && col < 2 * HS) bias = gate_b[col];
    #pragma unroll
    for (int e = 0; e < 8; ++e) {
      const int row = rowBase + e;
      float v = accs[f][e];
      if (MODE == 0) {
        outF[(size_t)row * Ncols + col] = v;
      } else {
        if (col < HS) {
          outA[(size_t)row * HS + col] = f2bf(sigmoidf_fast(v + bias));
        } else if (col < 2 * HS) {
          outI[(size_t)row * HS + (col - HS)] = f2bf(sigmoidf_fast(v + bias));
        } else {
          outV[(size_t)row * HS + (col - 2 * HS)] = f2bf(v);
        }
      }
    }
  }
}

// ---------------------------------------------------------------- chunked scan
__global__ void scan_pass1(const unsigned short* __restrict__ gA,
                           const unsigned short* __restrict__ gI,
                           const unsigned short* __restrict__ gV,
                           float* __restrict__ prodA,
                           float* __restrict__ lastH) {
  int tid  = blockIdx.x * blockDim.x + threadIdx.x;
  int lane = tid & (NLANES - 1);
  int c    = tid >> 12;
  if (c >= NC) return;
  int b = lane >> 10, j = lane & (HS - 1);
  size_t base = ((size_t)(b * SEQ + c * CL)) * HS + j;
  float p = 1.0f, h = 0.0f;
  for (int t = 0; t < CL; ++t, base += HS) {
    float a  = bf2f(gA[base]);
    float bv = bf2f(gI[base]) * bf2f(gV[base]);
    p *= a;
    h = a * h + bv;
  }
  prodA[(size_t)c * NLANES + lane] = p;
  lastH[(size_t)c * NLANES + lane] = h;
}

__global__ void scan_pass2(const float* __restrict__ prodA,
                           const float* __restrict__ lastH,
                           const float* __restrict__ h_prev,
                           float* __restrict__ carryIn,
                           float* __restrict__ hLastOut) {
  int lane = blockIdx.x * blockDim.x + threadIdx.x;
  if (lane >= NLANES) return;
  float h = h_prev[lane];
  for (int c = 0; c < NC; ++c) {
    carryIn[(size_t)c * NLANES + lane] = h;
    h = prodA[(size_t)c * NLANES + lane] * h + lastH[(size_t)c * NLANES + lane];
  }
  hLastOut[lane] = h;   // second output of the reference tuple
}

__global__ void scan_pass3(const unsigned short* __restrict__ gA,
                           const unsigned short* __restrict__ gI,
                           const unsigned short* __restrict__ gV,
                           const float* __restrict__ carryIn,
                           unsigned short* __restrict__ hseq) {
  int tid  = blockIdx.x * blockDim.x + threadIdx.x;
  int lane = tid & (NLANES - 1);
  int c    = tid >> 12;
  if (c >= NC) return;
  int b = lane >> 10, j = lane & (HS - 1);
  size_t base = ((size_t)(b * SEQ + c * CL)) * HS + j;
  float h = carryIn[(size_t)c * NLANES + lane];
  for (int t = 0; t < CL; ++t, base += HS) {
    float a  = bf2f(gA[base]);
    float bv = bf2f(gI[base]) * bf2f(gV[base]);
    h = a * h + bv;
    hseq[base] = f2bf(h);
  }
}

// ---------------------------------------------------------------- launch
extern "C" void kernel_launch(void* const* d_in, const int* in_sizes, int n_in,
                              void* d_out, int out_size, void* d_ws, size_t ws_size,
                              hipStream_t stream) {
  (void)in_sizes; (void)n_in; (void)out_size; (void)ws_size;
  const float* x       = (const float*)d_in[0];
  const float* h_prev  = (const float*)d_in[1];
  const float* gate_w  = (const float*)d_in[2];
  const float* gate_b  = (const float*)d_in[3];
  const float* value_w = (const float*)d_in[4];
  const float* out_w   = (const float*)d_in[5];
  float* y      = (float*)d_out;
  float* h_last = (float*)d_out + (size_t)MROWS * D_MODEL;

  // workspace carve-up (256B aligned)
  size_t off = 0;
  auto take = [&](size_t bytes) { size_t o = off; off += (bytes + 255) & ~(size_t)255; return o; };
  char* ws = (char*)d_ws;
  unsigned short* xb    = (unsigned short*)(ws + take((size_t)MROWS * D_MODEL * 2));
  unsigned short* w1b   = (unsigned short*)(ws + take((size_t)N1 * D_MODEL * 2));
  unsigned short* owb   = (unsigned short*)(ws + take((size_t)D_MODEL * HS * 2));
  unsigned short* gAb   = (unsigned short*)(ws + take((size_t)MROWS * HS * 2));
  unsigned short* gIb   = (unsigned short*)(ws + take((size_t)MROWS * HS * 2));
  unsigned short* gVb   = (unsigned short*)(ws + take((size_t)MROWS * HS * 2));
  unsigned short* hseq  = (unsigned short*)(ws + take((size_t)MROWS * HS * 2));
  float* prodA   = (float*)(ws + take((size_t)NC * NLANES * 4));
  float* lastH   = (float*)(ws + take((size_t)NC * NLANES * 4));
  float* carryIn = (float*)(ws + take((size_t)NC * NLANES * 4));

  // 1) f32 -> bf16 conversions
  int nx = MROWS * D_MODEL;
  f32_to_bf16_kernel<<<(nx + 255) / 256, 256, 0, stream>>>(x, xb, nx);
  int ng = NGATE * D_MODEL;
  f32_to_bf16_kernel<<<(ng + 255) / 256, 256, 0, stream>>>(gate_w, w1b, ng);
  int nv = HS * D_MODEL;
  f32_to_bf16_kernel<<<(nv + 255) / 256, 256, 0, stream>>>(value_w, w1b + (size_t)NGATE * D_MODEL, nv);
  int no = D_MODEL * HS;
  f32_to_bf16_kernel<<<(no + 255) / 256, 256, 0, stream>>>(out_w, owb, no);

  // 2) fused gate+value GEMM: [16384,1024] x [3072,1024]^T, sigmoid epilogue
  int blocks1 = (MROWS / BLOCK_M) * (N1 / BLOCK_N);   // 128 * 48 = 6144
  gla_gemm_bf16<1><<<blocks1, 256, 0, stream>>>(
      (const __bf16*)xb, (const __bf16*)w1b, N1,
      nullptr, gate_b, gAb, gIb, gVb);

  // 3) chunked parallel scan over T
  scan_pass1<<<(NLANES * NC) / 256, 256, 0, stream>>>(gAb, gIb, gVb, prodA, lastH);
  scan_pass2<<<NLANES / 256, 256, 0, stream>>>(prodA, lastH, h_prev, carryIn, h_last);
  scan_pass3<<<(NLANES * NC) / 256, 256, 0, stream>>>(gAb, gIb, gVb, carryIn, hseq);

  // 4) output projection: y = h_seq @ out_w^T
  int blocks2 = (MROWS / BLOCK_M) * (HS / BLOCK_N);   // 128 * 16 = 2048
  gla_gemm_bf16<0><<<blocks2, 256, 0, stream>>>(
      (const __bf16*)hseq, (const __bf16*)owb, HS,
      y, nullptr, nullptr, nullptr, nullptr);
}
